// MambaClassifier_50508815401012
// MI455X (gfx1250) — compile-verified, matrix operand order checked by reading, not run on hardware
//
#include <hip/hip_runtime.h>
#include <math.h>

typedef __attribute__((ext_vector_type(16))) __bf16 v16bf;
typedef __attribute__((ext_vector_type(8)))  float  v8f;

#define DEV __device__ __forceinline__

DEV __bf16 f2bf(float f) {
    union { float f; unsigned u; } a; a.f = f;
    unsigned r = a.u + 0x7FFFu + ((a.u >> 16) & 1u);   // round-to-nearest-even
    unsigned short h = (unsigned short)(r >> 16);
    return __builtin_bit_cast(__bf16, h);
}

DEV float gelu_exact(float x) {
    return 0.5f * x * (1.0f + erff(x * 0.70710678118654752f));
}

// Load one 16x32 bf16 WMMA A/B fragment (ISA 7.12.2 16-bit layout).
// p = row base already offset by lane-half K offset (kh).
DEV v16bf ldfrag(const __bf16* __restrict__ p) {
    v16bf f;
    #pragma unroll
    for (int v = 0; v < 8; ++v) {
        int kk = ((v & 3) << 1) + ((v >> 2) << 4);   // {0,2,4,6,16,18,20,22}
        f[2 * v]     = p[kk];
        f[2 * v + 1] = p[kk + 1];
    }
    return f;
}

// ---------------------------------------------------------------------------
// conv1: (B,L,8) -> (B,L,256), k=3 pad=1, ReLU.  One thread per output elem.
// ---------------------------------------------------------------------------
__global__ void k_conv1(const float* __restrict__ x, const float* __restrict__ w,
                        const float* __restrict__ b, float* __restrict__ out, int L) {
    int id = blockIdx.x * blockDim.x + threadIdx.x;         // < M*256
    int co = id & 255;
    int bt = id >> 8;
    int t  = bt % L, bb = bt / L;
    float acc = b[co];
    #pragma unroll
    for (int tap = 0; tap < 3; ++tap) {
        int tt = t + tap - 1;
        if (tt >= 0 && tt < L) {
            const float* xp = x + (size_t)(bb * L + tt) * 8;
            #pragma unroll
            for (int ci = 0; ci < 8; ++ci)
                acc += xp[ci] * w[(co * 8 + ci) * 3 + tap];
        }
    }
    out[id] = fmaxf(acc, 0.0f);
}

// ---------------------------------------------------------------------------
// im2col for conv2: (B,L,256) f32 -> (M, 768) bf16, k = tap*256+ci, zero pad.
// ---------------------------------------------------------------------------
__global__ void k_im2col(const float* __restrict__ src, __bf16* __restrict__ dst,
                         int L, int total) {
    int id = blockIdx.x * blockDim.x + threadIdx.x;
    if (id >= total) return;
    int k  = id % 768;
    int bt = id / 768;
    int tap = k >> 8, ci = k & 255;
    int t = bt % L, bb = bt / L;
    int tt = t + tap - 1;
    float v = (tt >= 0 && tt < L) ? src[(size_t)(bb * L + tt) * 256 + ci] : 0.0f;
    dst[id] = f2bf(v);
}

// conv2 weight repack: (co,ci,tap) f32 -> [co][tap*256+ci] bf16
__global__ void k_pack_w2(const float* __restrict__ w, __bf16* __restrict__ dst, int total) {
    int id = blockIdx.x * blockDim.x + threadIdx.x;
    if (id >= total) return;
    int co = id / 768, k = id % 768;
    int tap = k >> 8, ci = k & 255;
    dst[id] = f2bf(w[(co * 256 + ci) * 3 + tap]);
}

__global__ void k_f32_to_bf16(const float* __restrict__ src, __bf16* __restrict__ dst, int total) {
    int id = blockIdx.x * blockDim.x + threadIdx.x;
    if (id < total) dst[id] = f2bf(src[id]);
}

// ---------------------------------------------------------------------------
// WMMA GEMM:  C[M,N] (f32) = act( A[M,K](bf16) * W[N,K]^T(bf16) + bias ) [+ C]
// One wave computes a 32M x (16*NT)N tile: 2 A fragments x NT B fragments per
// K-step -> 2*NT v_wmma per iteration, no guards, no EXEC divergence.
// Requires: M % 32 == 0, N % (16*NT) == 0, K % 32 == 0.
// act: 0=none, 1=relu, 2=tanh.  acc: 1 -> C += result (residual path).
// ---------------------------------------------------------------------------
template<int NT>
__global__ void k_gemm_wmma(const __bf16* __restrict__ A, const __bf16* __restrict__ W,
                            const float* __restrict__ bias, float* __restrict__ C,
                            int M, int N, int K, int act, int acc) {
    int wave    = (blockIdx.x * blockDim.x + threadIdx.x) >> 5;
    int mGroups = M >> 5;                       // 32-row tiles
    int nGroups = N / (16 * NT);
    if (wave >= mGroups * nGroups) return;      // wave-uniform
    int mg = wave % mGroups;
    int ng = wave / mGroups;

    int lane = threadIdx.x & 31;
    int row  = lane & 15;          // M row (A) / N row (W) within tile
    int kh   = (lane >> 4) << 3;   // lane-half K offset (0 or 8)

    v8f accf[2][NT];
    #pragma unroll
    for (int mi = 0; mi < 2; ++mi)
        #pragma unroll
        for (int j = 0; j < NT; ++j)
            #pragma unroll
            for (int v = 0; v < 8; ++v) accf[mi][j][v] = 0.0f;

    const __bf16* aRow0 = A + ((size_t)(mg * 32) + row) * K + kh;
    const __bf16* aRow1 = aRow0 + (size_t)16 * K;
    const __bf16* bRow[NT];
    #pragma unroll
    for (int j = 0; j < NT; ++j)
        bRow[j] = W + ((size_t)(ng * 16 * NT + j * 16) + row) * K + kh;

    for (int k0 = 0; k0 < K; k0 += 32) {
        if (k0 + 32 < K) {
            __builtin_prefetch(aRow0 + k0 + 32, 0, 1);   // global_prefetch_b8
            __builtin_prefetch(aRow1 + k0 + 32, 0, 1);
        }
        v16bf af0 = ldfrag(aRow0 + k0);
        v16bf af1 = ldfrag(aRow1 + k0);
        #pragma unroll
        for (int j = 0; j < NT; ++j) {
            v16bf bf = ldfrag(bRow[j] + k0);
            accf[0][j] = __builtin_amdgcn_wmma_f32_16x16x32_bf16(
                false, af0, false, bf, (short)0, accf[0][j], false, false);
            accf[1][j] = __builtin_amdgcn_wmma_f32_16x16x32_bf16(
                false, af1, false, bf, (short)0, accf[1][j], false, false);
        }
    }

    int col = lane & 15;           // N within 16-tile (C/D layout)
    int mh  = (lane >> 4) << 3;    // M offset within 16-tile
    #pragma unroll
    for (int mi = 0; mi < 2; ++mi) {
        #pragma unroll
        for (int j = 0; j < NT; ++j) {
            int n0 = ng * 16 * NT + j * 16;
            float bv = bias ? bias[n0 + col] : 0.0f;
            #pragma unroll
            for (int v = 0; v < 8; ++v) {
                size_t idx = (size_t)(mg * 32 + mi * 16 + v + mh) * N + n0 + col;
                float o = accf[mi][j][v] + bv;
                if (act == 1)      o = fmaxf(o, 0.0f);
                else if (act == 2) o = tanhf(o);
                if (acc) o += C[idx];
                C[idx] = o;
            }
        }
    }
}

// ---------------------------------------------------------------------------
// Per-row LayerNorm over dm=256 -> bf16 output. One block (8 waves) per row.
// ---------------------------------------------------------------------------
__global__ void k_ln_bf16(const float* __restrict__ h, const float* __restrict__ g,
                          const float* __restrict__ b, __bf16* __restrict__ out) {
    int r = blockIdx.x, c = threadIdx.x;
    float v = h[(size_t)r * 256 + c];
    float s1 = v, s2 = v * v;
    #pragma unroll
    for (int o = 16; o > 0; o >>= 1) { s1 += __shfl_down(s1, o); s2 += __shfl_down(s2, o); }
    __shared__ float r1[8], r2[8], mu_s, rstd_s;
    int wid = c >> 5, lid = c & 31;
    if (lid == 0) { r1[wid] = s1; r2[wid] = s2; }
    __syncthreads();
    if (c == 0) {
        float t1 = 0, t2 = 0;
        #pragma unroll
        for (int i = 0; i < 8; ++i) { t1 += r1[i]; t2 += r2[i]; }
        float mu = t1 * (1.0f / 256.0f);
        float var = t2 * (1.0f / 256.0f) - mu * mu;
        mu_s = mu; rstd_s = rsqrtf(var + 1e-5f);
    }
    __syncthreads();
    out[(size_t)r * 256 + c] = f2bf((v - mu_s) * rstd_s * g[c] + b[c]);
}

// ---------------------------------------------------------------------------
// Depthwise conv (k=3,pad=1) over xp[:, :512] + exact GELU -> f32 + bf16.
// ---------------------------------------------------------------------------
__global__ void k_dwconv_gelu(const float* __restrict__ xp, const float* __restrict__ w,
                              const float* __restrict__ b, float* __restrict__ outf,
                              __bf16* __restrict__ outb, int L) {
    int id = blockIdx.x * blockDim.x + threadIdx.x;    // < M*512
    int d  = id & 511;
    int bt = id >> 9;
    int t = bt % L, bb = bt / L;
    float acc = b[d];
    #pragma unroll
    for (int tap = 0; tap < 3; ++tap) {
        int tt = t + tap - 1;
        if (tt >= 0 && tt < L)
            acc += xp[(size_t)(bb * L + tt) * 1024 + d] * w[d * 3 + tap];
    }
    float g = gelu_exact(acc);
    outf[id] = g;
    outb[id] = f2bf(g);
}

// ---------------------------------------------------------------------------
// SSM scan. One block per batch, 512 threads (thread = channel d).
// state[d, 0..15] in registers; per-step LN over the full (512,16) state via
// shuffle + LDS reduction; Bt/Ct staged through LDS.
// Writes y = out * gelu(x_res) in f32 (aliases xconv: same-index RMW) + bf16.
// ---------------------------------------------------------------------------
__global__ void k_scan(const float* __restrict__ bc,     // [B,L,32] tanh(Bp|Cp)
                       const float* __restrict__ xconv,  // [B,L,512]
                       const float* __restrict__ xres,   // xp base; res at col 512, ld 1024
                       const float* __restrict__ A, const float* __restrict__ D,
                       const float* __restrict__ sng, const float* __restrict__ snb,
                       float* __restrict__ yf, __bf16* __restrict__ yb, int L) {
    int bb = blockIdx.x, d = threadIdx.x;
    float st[16], tA[16], gv[16], bv[16];
    #pragma unroll
    for (int j = 0; j < 16; ++j) {
        st[j] = 0.0f;
        tA[j] = tanhf(A[d * 16 + j]);
        gv[j] = sng[d * 16 + j];
        bv[j] = snb[d * 16 + j];
    }
    float Dd = D[d];
    __shared__ float sBC[32];
    __shared__ float r1[16], r2[16], mu_s, rstd_s;
    int wid = d >> 5, lid = d & 31;

    for (int t = 0; t < L; ++t) {
        size_t bt = (size_t)bb * L + t;
        if (d < 32) sBC[d] = bc[bt * 32 + d];
        __syncthreads();

        float xt  = xconv[bt * 512 + d];
        float out = xt * Dd;
        #pragma unroll
        for (int j = 0; j < 16; ++j) {
            st[j] += sBC[j];             // state += Bt
            out   += st[j] * sBC[16 + j];
        }
        float rres = xres[bt * 1024 + 512 + d];
        float yv = out * gelu_exact(rres);
        yf[bt * 512 + d] = yv;
        yb[bt * 512 + d] = f2bf(yv);

        float s1 = 0.0f, s2 = 0.0f;
        #pragma unroll
        for (int j = 0; j < 16; ++j) {
            st[j] *= tA[j];              // state *= tanh(A)
            s1 += st[j];
            s2 += st[j] * st[j];
        }
        #pragma unroll
        for (int o = 16; o > 0; o >>= 1) { s1 += __shfl_down(s1, o); s2 += __shfl_down(s2, o); }
        if (lid == 0) { r1[wid] = s1; r2[wid] = s2; }
        __syncthreads();
        if (d == 0) {
            float t1 = 0, t2 = 0;
            #pragma unroll
            for (int i = 0; i < 16; ++i) { t1 += r1[i]; t2 += r2[i]; }
            float mu = t1 * (1.0f / 8192.0f);
            float var = t2 * (1.0f / 8192.0f) - mu * mu;
            mu_s = mu; rstd_s = rsqrtf(var + 1e-5f);
        }
        __syncthreads();
        float mu = mu_s, rstd = rstd_s;
        #pragma unroll
        for (int j = 0; j < 16; ++j)
            st[j] = (st[j] - mu) * rstd * gv[j] + bv[j];
        __syncthreads();                 // protect sBC for next step
    }
}

// ---------------------------------------------------------------------------
// Mean-pool over L + final FC (dm=256 -> nc=37). One block per batch.
// ---------------------------------------------------------------------------
__global__ void k_pool_fc(const float* __restrict__ h, const float* __restrict__ fw,
                          const float* __restrict__ fb, float* __restrict__ out, int L) {
    int bb = blockIdx.x, c = threadIdx.x;
    __shared__ float spool[256];
    float s = 0.0f;
    for (int t = 0; t < L; ++t) s += h[((size_t)bb * L + t) * 256 + c];
    spool[c] = s / (float)L;
    __syncthreads();
    if (c < 37) {
        float o = fb[c];
        #pragma unroll 8
        for (int dd = 0; dd < 256; ++dd) o += spool[dd] * fw[c * 256 + dd];
        out[bb * 37 + c] = o;
    }
}

// ===========================================================================
extern "C" void kernel_launch(void* const* d_in, const int* in_sizes, int n_in,
                              void* d_out, int out_size, void* d_ws, size_t ws_size,
                              hipStream_t stream) {
    (void)in_sizes; (void)n_in; (void)out_size; (void)ws_size;
    const int B = 128, L = 400, nl = 2;
    const int M = B * L;                  // 51200

    const float* x        = (const float*)d_in[0];
    const float* conv1_w  = (const float*)d_in[1];
    const float* conv1_b  = (const float*)d_in[2];
    const float* conv2_w  = (const float*)d_in[3];
    const float* conv2_b  = (const float*)d_in[4];
    const float* ln_g     = (const float*)d_in[5];
    const float* ln_b     = (const float*)d_in[6];
    const float* inproj_w = (const float*)d_in[7];
    const float* inproj_b = (const float*)d_in[8];
    const float* dw_w     = (const float*)d_in[9];
    const float* dw_b     = (const float*)d_in[10];
    const float* xproj_w  = (const float*)d_in[11];
    const float* xproj_b  = (const float*)d_in[12];
    const float* A_in     = (const float*)d_in[13];
    const float* D_in     = (const float*)d_in[14];
    const float* sn_g     = (const float*)d_in[15];
    const float* sn_b     = (const float*)d_in[16];
    const float* outproj_w= (const float*)d_in[17];
    const float* outproj_b= (const float*)d_in[18];
    const float* fc_w     = (const float*)d_in[19];
    const float* fc_b     = (const float*)d_in[20];
    float* outp = (float*)d_out;

    // ---- carve workspace ----
    char* ws = (char*)d_ws;
    size_t cur = 0;
    auto take = [&](size_t bytes) -> char* {
        char* p = ws + cur;
        cur += (bytes + 255) & ~(size_t)255;
        return p;
    };
    __bf16* wpk2   = (__bf16*)take((size_t)256 * 768 * 2);   // conv2 weight, packed
    __bf16* w_in   = (__bf16*)take((size_t)1024 * 256 * 2);  // inproj weight (per layer)
    __bf16* w_x    = (__bf16*)take((size_t)32 * 512 * 2);    // xproj weight
    __bf16* w_out  = (__bf16*)take((size_t)256 * 512 * 2);   // outproj weight
    float*  f_h    = (float*) take((size_t)M * 256 * 4);     // residual stream
    float*  f_t0   = (float*) take((size_t)M * 256 * 4);     // conv1 output
    __bf16* abf    = (__bf16*)take((size_t)M * 768 * 2);     // bf16 A operand (reused)
    float*  f_xp   = (float*) take((size_t)M * 1024 * 4);    // inproj output
    float*  f_xc   = (float*) take((size_t)M * 512 * 4);     // gelu(dwconv); aliased as y
    float*  f_bc   = (float*) take((size_t)M * 32 * 4);      // tanh(Bp|Cp)

    const int mG = M / 32;                                    // 1600 32-row tiles

    // ---- front-end convs ----
    k_conv1<<<(M * 256) / 256, 256, 0, stream>>>(x, conv1_w, conv1_b, f_t0, L);
    k_im2col<<<(M * 768) / 256, 256, 0, stream>>>(f_t0, abf, L, M * 768);
    k_pack_w2<<<(256 * 768) / 256, 256, 0, stream>>>(conv2_w, wpk2, 256 * 768);
    // conv2 as GEMM: M x 256, K=768, ReLU -> residual stream h
    k_gemm_wmma<4><<<mG * (256 / 64) / 8, 256, 0, stream>>>(abf, wpk2, conv2_b, f_h,
                                                            M, 256, 768, /*act*/1, /*acc*/0);

    // ---- Mamba layers ----
    for (int l = 0; l < nl; ++l) {
        // LayerNorm -> bf16 A operand
        k_ln_bf16<<<M, 256, 0, stream>>>(f_h, ln_g + l * 256, ln_b + l * 256, abf);
        // inproj: [M,256] x [1024,256]^T
        k_f32_to_bf16<<<(1024 * 256) / 256, 256, 0, stream>>>(inproj_w + (size_t)l * 1024 * 256,
                                                              w_in, 1024 * 256);
        k_gemm_wmma<4><<<mG * (1024 / 64) / 8, 256, 0, stream>>>(abf, w_in, inproj_b + l * 1024,
                                                                 f_xp, M, 1024, 256, 0, 0);
        // depthwise conv + GELU on x_conv half
        k_dwconv_gelu<<<(M * 512) / 256, 256, 0, stream>>>(f_xp, dw_w + l * 512 * 3,
                                                           dw_b + l * 512, f_xc, abf, L);
        // xproj: [M,512] x [32,512]^T, tanh fused -> Bp|Cp
        k_f32_to_bf16<<<(32 * 512) / 256, 256, 0, stream>>>(xproj_w + (size_t)l * 32 * 512,
                                                            w_x, 32 * 512);
        k_gemm_wmma<2><<<mG * (32 / 32) / 8, 256, 0, stream>>>(abf, w_x, xproj_b + l * 32,
                                                               f_bc, M, 32, 512, /*tanh*/2, 0);
        // sequential scan (writes y in-place over f_xc + bf16 copy into abf)
        k_scan<<<B, 512, 0, stream>>>(f_bc, f_xc, f_xp,
                                      A_in + (size_t)l * 512 * 16, D_in + l * 512,
                                      sn_g + (size_t)l * 512 * 16, sn_b + (size_t)l * 512 * 16,
                                      f_xc, abf, L);
        // outproj: [M,512] x [256,512]^T, accumulate into residual h
        k_f32_to_bf16<<<(256 * 512) / 256, 256, 0, stream>>>(outproj_w + (size_t)l * 256 * 512,
                                                             w_out, 256 * 512);
        k_gemm_wmma<4><<<mG * (256 / 64) / 8, 256, 0, stream>>>(abf, w_out, outproj_b + l * 256,
                                                                f_h, M, 256, 512, 0, /*acc*/1);
    }

    // ---- pool + classifier ----
    k_pool_fc<<<B, 256, 0, stream>>>(f_h, fc_w, fc_b, outp, L);
}
